// TriangularUpdate_71012989272671
// MI455X (gfx1250) — compile-verified, matrix operand order checked by reading, not run on hardware
//
#include <hip/hip_runtime.h>
#include <hip/hip_bf16.h>

// ---------------------------------------------------------------------------
// TriangularUpdate for MI455X (gfx1250): bf16 WMMA pipeline.
//   K1: LayerNorm  (fp32 -> bf16)
//   K2: proj V/E + sigmoid gating (WMMA), writes 4 gated tensors in GEMM-ready
//       layouts (transposes absorbed in a branchless epilogue scatter)
//   K3: 128 batched 512x512x512 GEMMs (v_wmma_f32_16x16x32_bf16).
//       A-panel staged to LDS by the Tensor Data Mover (tensor_load_to_lds,
//       TDM row-padding used for LDS bank staggering) when the builtin exists;
//       manual global->LDS staging otherwise.
//   K4: out-projection + sigmoid gate (WMMA), fp32 output
// ---------------------------------------------------------------------------

#define BQ   2
#define NQ   512
#define EWQ  64
#define HQ   32
#define NN   (NQ * NQ)                 // 262144 = 2^18
#define MEDGE ((long)BQ * NN)          // 524288 edges
#define GATED_ELEMS ((long)BQ * HQ * NQ * NQ)   // 16,777,216 per tensor

#if __has_builtin(__builtin_amdgcn_tensor_load_to_lds) && \
    __has_builtin(__builtin_amdgcn_s_wait_tensorcnt)
#define HAVE_TDM 1
#else
#define HAVE_TDM 0
#endif

typedef __attribute__((ext_vector_type(16))) __bf16 v16bf;
typedef __attribute__((ext_vector_type(8)))  float  v8f;
typedef __attribute__((ext_vector_type(4)))  unsigned int u32x4;
typedef __attribute__((ext_vector_type(2)))  float  f32x2;
typedef __attribute__((ext_vector_type(4))) unsigned int tdm_g0_t;
typedef __attribute__((ext_vector_type(8))) int          tdm_g1_t;
typedef __attribute__((ext_vector_type(4))) int          tdm_g2_t;

union BFrag { u32x4 u[2]; v16bf v; };

__device__ __forceinline__ unsigned short f32_to_bf16(float f) {
  union { float f; unsigned int u; } x; x.f = f;
  unsigned int r = x.u + 0x7FFFu + ((x.u >> 16) & 1u);   // round-nearest-even
  return (unsigned short)(r >> 16);
}

// Fast sigmoid: v_exp_f32 + v_rcp_f32 (avoid the IEEE div expansion; ~1 ulp
// rcp is far below the bf16/gating precision of this layer).
__device__ __forceinline__ float sigmoidf(float x) {
  return __builtin_amdgcn_rcpf(1.0f + __expf(-x));
}

// B-operand swizzle: position of K index k (0..31) inside a lane-contiguous
// 32-half run, mirroring the documented 16-bit A-matrix K ordering:
// half = (k>>3)&1 ; within half K order is {0..7,16..23} packed pairwise.
__device__ __forceinline__ int bperm(int k) {
  return (((k >> 3) & 1) << 4) + ((k & 16) >> 1) + (k & 7);
}

// A fragment (16x32 bf16) from LDS; rows at stride rs halfs (rs % 8 == 0 for
// 16B-aligned ds_load_b128, chosen so 16 consecutive rows hit distinct banks).
// ISA layout: lane L(0-15): M=L, K in {0..7,16..23}; lane L+16: K in {8..15,24..31}.
__device__ __forceinline__ v16bf load_a_frag(const unsigned short* lds, int m0,
                                             int rs, int k0, int lane) {
  int M = lane & 15, hf = lane >> 4;
  const unsigned short* p = lds + (m0 + M) * rs + k0;
  BFrag f;
  f.u[0] = *(const u32x4*)(p + hf * 8);
  f.u[1] = *(const u32x4*)(p + 16 + hf * 8);
  return f.v;
}

// B fragment (32x16 bf16) from pre-swizzled LDS block: [n(16)][40] halfs,
// lane's 16 values are contiguous (two ds_load_b128).
__device__ __forceinline__ v16bf load_b_frag(const unsigned short* base, int lane) {
  int n = lane & 15, hf = lane >> 4;
  const unsigned short* p = base + n * 40 + hf * 16;
  BFrag f;
  f.u[0] = *(const u32x4*)(p);
  f.u[1] = *(const u32x4*)(p + 8);
  return f.v;
}

__device__ __forceinline__ v8f wmma_bf16(v16bf a, v16bf b, v8f c) {
  return __builtin_amdgcn_wmma_f32_16x16x32_bf16(
      /*neg_a=*/false, a, /*neg_b=*/false, b,
      /*c_mod=*/(short)0, c, /*reuse_a=*/false, /*reuse_b=*/false);
}

#if HAVE_TDM
// TDM: load a 128-row x 128-half 2D tile of a 512x512 bf16 tensor into LDS,
// inserting 16B (4 DWORD) padding after every 256B (64 DWORD) row so the LDS
// image has row stride 136 halfs (bank-staggered, 16B aligned).
// D# field packing per CDNA5 ISA.
__device__ __forceinline__ void tdm_load_a_slab(const unsigned short* gsrc,
                                                unsigned int lds_off) {
  unsigned long long ga = (unsigned long long)(size_t)gsrc;
  tdm_g0_t g0;
  g0[0] = 1u;                                   // count=1, user descriptor
  g0[1] = lds_off;                              // lds_addr (bytes)
  g0[2] = (unsigned int)(ga & 0xFFFFFFFFu);     // global_addr[31:0]
  g0[3] = (unsigned int)((ga >> 32) & 0x1FFFFFFu) | (2u << 30); // [56:32]+type=2
  tdm_g1_t g1;
  g1[0] = (1 << 16)      // data_size = 2 bytes
        | (1 << 20)      // pad_enable
        | (5 << 22)      // pad_interval: 64 DWORDs (=256B = 128 halfs)
        | (3 << 25);     // pad_amount: 4 DWORDs (=16B = 8 halfs)
  g1[1] = (int)((NQ & 0xFFFF) << 16);           // tensor_dim0[15:0] @ bits63:48
  g1[2] = (int)((NQ & 0xFFFF) << 16);           // tensor_dim1[15:0] @ bits95:80
  g1[3] = (int)(128u << 16);                    // tile_dim0 = 128 @ bits127:112
  g1[4] = 128;                                  // tile_dim1 = 128 @ bits143:128
  g1[5] = NQ;                                   // tensor_dim0_stride[31:0]
  g1[6] = 0;
  g1[7] = 0;
  tdm_g2_t z4 = {0, 0, 0, 0};
#if __clang_major__ >= 23
  tdm_g1_t z8 = {0, 0, 0, 0, 0, 0, 0, 0};
  __builtin_amdgcn_tensor_load_to_lds(g0, g1, z4, z4, z8, 0);
#else
  __builtin_amdgcn_tensor_load_to_lds(g0, g1, z4, z4, 0);
#endif
}
#endif

// ---------------------------------------------------------------------------
// K1: LayerNorm over ew=64, one wave32 per edge (2 values/lane), bf16 out.
// ---------------------------------------------------------------------------
__global__ __launch_bounds__(256) void ln_kernel(
    const float* __restrict__ e, const float* __restrict__ g,
    const float* __restrict__ b, unsigned short* __restrict__ eln) {
  long row = (long)blockIdx.x * 8 + (threadIdx.x >> 5);
  int lane = threadIdx.x & 31;
  int c = lane * 2;
  f32x2 v = *(const f32x2*)(e + row * EWQ + c);
  float s = v.x + v.y;
  float sq = v.x * v.x + v.y * v.y;
#pragma unroll
  for (int off = 16; off >= 1; off >>= 1) {
    s  += __shfl_xor(s, off, 32);
    sq += __shfl_xor(sq, off, 32);
  }
  float mean = s * (1.0f / 64.0f);
  float var  = sq * (1.0f / 64.0f) - mean * mean;
  float inv  = rsqrtf(var + 1e-5f);
  float o0 = (v.x - mean) * inv * g[c]     + b[c];
  float o1 = (v.y - mean) * inv * g[c + 1] + b[c + 1];
  unsigned int packed = (unsigned int)f32_to_bf16(o0) |
                        ((unsigned int)f32_to_bf16(o1) << 16);
  *(unsigned int*)(eln + row * EWQ + c) = packed;
}

// ---------------------------------------------------------------------------
// K2: [64-edge tile, 64] x [64, 256(=WV||WE)] WMMA GEMM + bias + mask +
//     sigmoid gating. Writes 4 gated bf16 tensors:
//       tensor 0: V_in  at [b,h,c,r] (transposed -> B-operand [k][j] for "in")
//       tensor 1: V_out at [b,h,r,c] (natural    -> B-operand for "out")
//       tensor 2: E_in  at [b,h,r,c] (natural    -> A-operand [i][k] for "in")
//       tensor 3: E_out at [b,h,c,r] (transposed -> A-operand for "out")
// Epilogue scatter is branchless: transposed vs natural is just a swap of the
// row/col multipliers, chosen once per wave.
// ---------------------------------------------------------------------------
__global__ __launch_bounds__(256) void proj_gate_kernel(
    const unsigned short* __restrict__ eln, const float* __restrict__ mask,
    const float* __restrict__ W_V, const float* __restrict__ b_V,
    const float* __restrict__ W_E, const float* __restrict__ b_E,
    unsigned short* __restrict__ gws) {
  __shared__ __attribute__((aligned(16))) unsigned short sA[64 * 72];
  __shared__ __attribute__((aligned(16))) unsigned short sW[32 * 16 * 40];
  __shared__ float sBias[256];
  __shared__ float sMask[64];

  const int tid  = threadIdx.x;
  const int lane = tid & 31;
  const int wave = tid >> 5;
  const long mbase = (long)blockIdx.x * 64;

  // Stage weights (64 x 256 fp32) -> bf16, pre-swizzled into B-fragment layout:
  // block (tn, ks) holds [n(16)][40] with K position bperm(k&31).
  for (int idx = tid; idx < 64 * 256; idx += 256) {
    int k = idx >> 8, c = idx & 255;
    float w = (c < 128) ? W_V[k * 128 + c] : W_E[k * 128 + (c - 128)];
    int tn = c >> 4, n = c & 15, ks = k >> 5, kl = k & 31;
    sW[((tn * 2 + ks) * 16 + n) * 40 + bperm(kl)] = f32_to_bf16(w);
  }
  sBias[tid] = (tid < 128) ? b_V[tid] : b_E[tid - 128];
  if (tid < 64) sMask[tid] = mask[mbase + tid];
  // Stage A: 64 edges x 64 bf16, padded rows (72 halfs) for bank spread.
  for (int idx = tid; idx < 512; idx += 256) {
    int row = idx >> 3, part = idx & 7;
    *(u32x4*)(&sA[row * 72 + part * 8]) =
        *(const u32x4*)(eln + (mbase + row) * EWQ + part * 8);
  }
  __syncthreads();

  // Wave -> gated pair: tensor id t = wave>>1 (0=Vin,1=Vout,2=Ein,3=Eout),
  // head-half = wave&1. Gate GEMM tile tg, lin tile tl = tg+2.
  const int tens = wave >> 1;
  const int hoff = (wave & 1) * 16;
  const int tg = tens * 4 + (wave & 1);
  const int tl = tg + 2;
  const bool transposed = (tens == 0) || (tens == 3);   // Vin, Eout
  unsigned short* dst = gws + (long)tens * GATED_ELEMS;

  const int n  = lane & 15;
  const int hf = lane >> 4;
  const float bg = sBias[tg * 16 + n];
  const float bl = sBias[tl * 16 + n];
  const int h = hoff + n;

  // Branchless scatter setup: all 64 edges in this block share batch index bb
  // (64 divides N*N); transposed layout just swaps the (r,c) multipliers.
  const int rm = transposed ? 1 : NQ;
  const int cm = transposed ? NQ : 1;
  const int bb = (int)(mbase >> 18);
  const int rembase = (int)(mbase & (NN - 1));
  const long tbase = (long)(bb * HQ + h) * NN;

#pragma unroll
  for (int msub = 0; msub < 4; ++msub) {
    v8f cg = {0, 0, 0, 0, 0, 0, 0, 0};
    v8f cl = {0, 0, 0, 0, 0, 0, 0, 0};
#pragma unroll
    for (int ks = 0; ks < 2; ++ks) {
      v16bf a = load_a_frag(sA, msub * 16, 72, ks * 32, lane);
      v16bf bgf = load_b_frag(&sW[((tg * 2 + ks) * 16) * 40], lane);
      v16bf blf = load_b_frag(&sW[((tl * 2 + ks) * 16) * 40], lane);
      cg = wmma_bf16(a, bgf, cg);
      cl = wmma_bf16(a, blf, cl);
    }
#pragma unroll
    for (int v = 0; v < 8; ++v) {
      int m = msub * 16 + v + 8 * hf;
      int rem = rembase + m;
      int r = rem >> 9, c = rem & (NQ - 1);
      float val = sigmoidf(cg[v] + bg + sMask[m]) * (cl[v] + bl);
      dst[tbase + r * rm + c * cm] = f32_to_bf16(val);
    }
  }
}

// ---------------------------------------------------------------------------
// K3: 128 batched C[i,j] = sum_k A[i,k] * B[k,j] (512^3), bf16 WMMA.
// Block: 256 threads / 8 waves, macro-tile 128(i) x 64(j).
// A: 128x128-half K-slab in LDS (TDM with pad, or manual), stride 136 halfs.
// B: 32x64 panel per K-step, pre-swizzled to fragment layout.
// Wave w: i-group = w>>2 (4 i-tiles), j-tile = w&3  -> 4 WMMAs per B frag.
// Slab loop rolled, k-step loop unrolled: TDM issue/wait and barriers are
// compile-time placed (no divergent branches in the hot loop).
// ---------------------------------------------------------------------------
__global__ __launch_bounds__(256) void triangle_kernel(
    const unsigned short* __restrict__ gws, unsigned short* __restrict__ va) {
  __shared__ __attribute__((aligned(16))) unsigned short sAa[128 * 136];
  __shared__ __attribute__((aligned(16))) unsigned short sBb[64 * 40];

  const int tid = threadIdx.x, lane = tid & 31, wave = tid >> 5;
  const int i0 = blockIdx.x * 128;        // gridDim.x = 4
  const int j0 = blockIdx.y * 64;         // gridDim.y = 8
  const int batch = blockIdx.z;           // which*(B*H) + b*H + h
  const int which = batch >> 6;
  const int bh = batch & 63;
  const int b = bh >> 5, h = bh & 31;

  const unsigned short* Aptr =
      gws + (long)(which ? 3 : 2) * GATED_ELEMS + (long)(b * HQ + h) * NN;
  const unsigned short* Bptr =
      gws + (long)(which ? 1 : 0) * GATED_ELEMS + (long)(b * HQ + h) * NN;

  const int ig = wave >> 2;               // i-tile group: tiles ig*4 + 0..3
  const int jt = wave & 3;                // j-subtile

  v8f acc[4];
#pragma unroll
  for (int t = 0; t < 4; ++t) acc[t] = (v8f){0, 0, 0, 0, 0, 0, 0, 0};

  for (int slab = 0; slab < 4; ++slab) {
    __syncthreads();                      // prior slab's sAa/sBb readers done
#if HAVE_TDM
    if (wave == 0)
      tdm_load_a_slab(Aptr + (long)i0 * NQ + slab * 128,
                      (unsigned int)(size_t)sAa);
#else
    for (int idx = tid; idx < 2048; idx += 256) {
      int row = idx >> 4, part = idx & 15;
      *(u32x4*)(&sAa[row * 136 + part * 8]) =
          *(const u32x4*)(Aptr + (long)(i0 + row) * NQ + slab * 128 + part * 8);
    }
#endif
#pragma unroll
    for (int kki = 0; kki < 4; ++kki) {
      const int kk = slab * 4 + kki;
      if (kki) __syncthreads();           // previous step's sBb readers done
      // Stage B panel 32(k) x 64(j), pre-swizzled to fragment layout [j][40].
      for (int idx = tid; idx < 2048; idx += 256) {
        int k = idx >> 6, j = idx & 63;
        sBb[j * 40 + bperm(k)] = Bptr[(long)(kk * 32 + k) * NQ + j0 + j];
        if (kk < 15)   // prefetch next K panel -> global_prefetch
          __builtin_prefetch(Bptr + (long)(kk * 32 + 32 + k) * NQ + j0 + j,
                             0, 1);
      }
#if HAVE_TDM
      if (kki == 0 && wave == 0) __builtin_amdgcn_s_wait_tensorcnt(0);
#endif
      __syncthreads();

      v16bf bf = load_b_frag(&sBb[(jt * 16) * 40], lane);
#pragma unroll
      for (int t = 0; t < 4; ++t) {
        v16bf a = load_a_frag(sAa, (ig * 4 + t) * 16, 136, kki * 32, lane);
        acc[t] = wmma_bf16(a, bf, acc[t]);
      }
    }
  }

  // Epilogue: Va[b,i,j, ch] bf16, ch = which*H + h.
  const int n = lane & 15, hf = lane >> 4;
  const int ch = which * HQ + h;
  const int j = j0 + jt * 16 + n;
#pragma unroll
  for (int t = 0; t < 4; ++t) {
#pragma unroll
    for (int v = 0; v < 8; ++v) {
      int i = i0 + (ig * 4 + t) * 16 + v + 8 * hf;
      va[(((long)b * NQ + i) * NQ + j) * EWQ + ch] = f32_to_bf16(acc[t][v]);
    }
  }
}

// ---------------------------------------------------------------------------
// K4: [64-edge tile, 64] x [64, 128] WMMA GEMM + bias, out = sig(g)*l, fp32.
// Wave w: out tile t=w&3 (gate cols t*16, lin cols 64+t*16), m-subtiles
// {2*(w>>2), 2*(w>>2)+1}.
// ---------------------------------------------------------------------------
__global__ __launch_bounds__(256) void out_kernel(
    const unsigned short* __restrict__ va, const float* __restrict__ W_O,
    const float* __restrict__ b_O, float* __restrict__ out) {
  __shared__ __attribute__((aligned(16))) unsigned short sA[64 * 72];
  __shared__ __attribute__((aligned(16))) unsigned short sW[16 * 16 * 40];
  __shared__ float sBias[128];

  const int tid = threadIdx.x, lane = tid & 31, wave = tid >> 5;
  const long mbase = (long)blockIdx.x * 64;

  for (int idx = tid; idx < 64 * 128; idx += 256) {
    int k = idx >> 7, c = idx & 127;
    int tn = c >> 4, n = c & 15, ks = k >> 5, kl = k & 31;
    sW[((tn * 2 + ks) * 16 + n) * 40 + bperm(kl)] = f32_to_bf16(W_O[idx]);
  }
  if (tid < 128) sBias[tid] = b_O[tid];
  for (int idx = tid; idx < 512; idx += 256) {
    int row = idx >> 3, part = idx & 7;
    *(u32x4*)(&sA[row * 72 + part * 8]) =
        *(const u32x4*)(va + (mbase + row) * EWQ + part * 8);
  }
  __syncthreads();

  const int t = wave & 3;
  const int tg = t, tl = t + 4;
  const int n = lane & 15, hf = lane >> 4;
  const float bg = sBias[tg * 16 + n];
  const float bl = sBias[64 + t * 16 + n];

#pragma unroll
  for (int mi = 0; mi < 2; ++mi) {
    int msub = (wave >> 2) * 2 + mi;
    v8f cg = {0, 0, 0, 0, 0, 0, 0, 0};
    v8f cl = {0, 0, 0, 0, 0, 0, 0, 0};
#pragma unroll
    for (int ks = 0; ks < 2; ++ks) {
      v16bf a = load_a_frag(sA, msub * 16, 72, ks * 32, lane);
      v16bf bgf = load_b_frag(&sW[((tg * 2 + ks) * 16) * 40], lane);
      v16bf blf = load_b_frag(&sW[((tl * 2 + ks) * 16) * 40], lane);
      cg = wmma_bf16(a, bgf, cg);
      cl = wmma_bf16(a, blf, cl);
    }
#pragma unroll
    for (int v = 0; v < 8; ++v) {
      int m = msub * 16 + v + 8 * hf;
      out[(mbase + m) * EWQ + t * 16 + n] = sigmoidf(cg[v] + bg) * (cl[v] + bl);
    }
  }
}

// ---------------------------------------------------------------------------
extern "C" void kernel_launch(void* const* d_in, const int* in_sizes, int n_in,
                              void* d_out, int out_size, void* d_ws,
                              size_t ws_size, hipStream_t stream) {
  const float* e    = (const float*)d_in[0];
  const float* mask = (const float*)d_in[1];
  const float* ln_g = (const float*)d_in[2];
  const float* ln_b = (const float*)d_in[3];
  const float* W_V  = (const float*)d_in[4];
  const float* b_V  = (const float*)d_in[5];
  const float* W_E  = (const float*)d_in[6];
  const float* b_E  = (const float*)d_in[7];
  const float* W_O  = (const float*)d_in[8];
  const float* b_O  = (const float*)d_in[9];
  float* out = (float*)d_out;

  // ws: [4 x GATED_ELEMS bf16 gated tensors][MEDGE*64 bf16 scratch]
  // scratch holds e_ln for K2, then is reused for Va by K3/K4 (stream-ordered).
  unsigned short* gws     = (unsigned short*)d_ws;
  unsigned short* scratch = gws + 4 * GATED_ELEMS;

  ln_kernel<<<(int)(MEDGE / 8), 256, 0, stream>>>(e, ln_g, ln_b, scratch);
  proj_gate_kernel<<<(int)(MEDGE / 64), 256, 0, stream>>>(
      scratch, mask, W_V, b_V, W_E, b_E, gws);
  dim3 g3(NQ / 128, NQ / 64, 2 * BQ * HQ);   // (4, 8, 128)
  triangle_kernel<<<g3, 256, 0, stream>>>(gws, scratch);
  out_kernel<<<(int)(MEDGE / 64), 256, 0, stream>>>(scratch, W_O, b_O, out);
}
